// hDRMLPv7Embed_70076686402006
// MI455X (gfx1250) — matrix operand order repaired and checked by reading, not compiled, not run on hardware
//
#include <hip/hip_runtime.h>
#include <hip/hip_bf16.h>
#include <math.h>

// ---------------------------------------------------------------------------
// CDNA5 / gfx1250 implementation of HierarchicalMultiScaleRegionLayer v4.
//   K_cvt    : fp32 -> bf16 weight conversion (OIHW flat == GEMM K order)
//   K_first  : 3x3 conv (3->24)+BN+GELU -> gh (fp32) AND A1 cols 0..383 (bf16)
//   K_branch1: tiled dw3x3+BN+GELU+1x1+BN+1x1res -> gelu -> gb1 AND A1 384..575
//   K_branch2: tiled dw3x3+..., identity res -> gelu -> A1 cols 576..767 only
//   K_gemm<0>: A1 x W1^T (WMMA bf16) + BN + GELU -> scatter A2 (im2col conv2)
//   K_gemm<1>: A2 x W2^T + BN + GELU -> scatter A3 (im2col conv3)
//   K_gemm<2>: A3 x W3^T + BN -> d_out [B,196,768] fp32
// GEMM: whole 64x768 B panel staged in LDS once via async-to-LDS (ASYNCcnt),
// barrier-free K loop, padded LDS stride for bank-conflict-free fragments.
// ---------------------------------------------------------------------------

typedef __attribute__((ext_vector_type(16))) __bf16 v16bf;
typedef __attribute__((ext_vector_type(8)))  __bf16 v8bf;
typedef __attribute__((ext_vector_type(8)))  float  v8f;
typedef __attribute__((ext_vector_type(4)))  int    v4i;

#define BN_EPS 1e-5f
#define AS1 __attribute__((address_space(1)))
#define AS3 __attribute__((address_space(3)))

#if defined(__HIP_DEVICE_COMPILE__) && __has_builtin(__builtin_amdgcn_global_load_async_to_lds_b128)
#define USE_ASYNC_LDS 1
#else
#define USE_ASYNC_LDS 0
#endif

__device__ __forceinline__ float geluf(float x) {
  return 0.5f * x * (1.0f + erff(x * 0.70710678118654752f));
}

__device__ __forceinline__ void wait_async0() {
#if defined(__HIP_DEVICE_COMPILE__)
#if __has_builtin(__builtin_amdgcn_s_wait_asynccnt)
  __builtin_amdgcn_s_wait_asynccnt(0);
#else
  asm volatile("s_wait_asynccnt 0x0" ::: "memory");
#endif
#endif
}

// ---------------------------------------------------------------------------
// fp32 -> bf16 elementwise
// ---------------------------------------------------------------------------
__global__ void cvt_bf16_kernel(const float* __restrict__ src,
                                __bf16* __restrict__ dst, int n) {
  int i = blockIdx.x * 256 + threadIdx.x;
  if (i < n) dst[i] = (__bf16)src[i];
}

// ---------------------------------------------------------------------------
// First conv 3x3 SAME 3->24 + BN + GELU.
// Writes gh fp32 AND bf16 slice of A1 (channels 0..23).
// block (32,8), grid (7,28,32)
// ---------------------------------------------------------------------------
__global__ void first_conv_kernel(const float* __restrict__ x,
                                  const float* __restrict__ w,    // [24,3,3,3]
                                  const float* __restrict__ bg,
                                  const float* __restrict__ bb,
                                  const float* __restrict__ bm,
                                  const float* __restrict__ bv,
                                  float* __restrict__ gh,
                                  __bf16* __restrict__ A1) {
  __shared__ float sw[24 * 27];
  __shared__ float sinv[24], soff[24];
  int tid = threadIdx.y * 32 + threadIdx.x;
  for (int i = tid; i < 24 * 27; i += 256) sw[i] = w[i];
  if (tid < 24) {
    float inv = bg[tid] * rsqrtf(bv[tid] + BN_EPS);
    sinv[tid] = inv;
    soff[tid] = bb[tid] - bm[tid] * inv;
  }
  __syncthreads();

  int x0 = blockIdx.x * 32 + threadIdx.x;
  int y0 = blockIdx.y * 8 + threadIdx.y;
  int b  = blockIdx.z;

  float patch[3][9];
  #pragma unroll
  for (int ci = 0; ci < 3; ++ci)
    #pragma unroll
    for (int dy = -1; dy <= 1; ++dy)
      #pragma unroll
      for (int dx = -1; dx <= 1; ++dx) {
        int yy = y0 + dy, xx = x0 + dx;
        float v = 0.f;
        if (yy >= 0 && yy < 224 && xx >= 0 && xx < 224)
          v = x[(((size_t)b * 3 + ci) * 224 + yy) * 224 + xx];
        patch[ci][(dy + 1) * 3 + (dx + 1)] = v;
      }

  // im2col target row/col pieces (4x4 non-overlapping patches)
  size_t m = (size_t)b * 3136 + (y0 >> 2) * 56 + (x0 >> 2);
  int colbase = (y0 & 3) * 4 + (x0 & 3);

  for (int co = 0; co < 24; ++co) {
    float acc = 0.f;
    #pragma unroll
    for (int ci = 0; ci < 3; ++ci)
      #pragma unroll
      for (int k = 0; k < 9; ++k) acc += patch[ci][k] * sw[co * 27 + ci * 9 + k];
    float g = geluf(acc * sinv[co] + soff[co]);
    gh[(((size_t)b * 24 + co) * 224 + y0) * 224 + x0] = g;
    A1[m * 768 + co * 16 + colbase] = (__bf16)g;
  }
}

// ---------------------------------------------------------------------------
// Branch1: 8x8 grid of 28x28 tiles, cin=24, cout=12, 1x1 residual (v2).
// Writes gb1 fp32 AND A1 slice (channels 24..35). block (28,4), grid (8,56,32)
// ---------------------------------------------------------------------------
__global__ void branch1_kernel(const float* __restrict__ gh,
                               const float* __restrict__ wdw,   // [64,24,9]
                               const float* __restrict__ g1, const float* __restrict__ b1,
                               const float* __restrict__ m1, const float* __restrict__ v1,
                               const float* __restrict__ wpw,   // [64,12,24]
                               const float* __restrict__ g2, const float* __restrict__ b2,
                               const float* __restrict__ m2, const float* __restrict__ v2,
                               const float* __restrict__ wres,  // [64,12,24]
                               float* __restrict__ gb1,
                               __bf16* __restrict__ A1) {
  __shared__ float sdw[24 * 9];
  __shared__ float sinv1[24], soff1[24];
  __shared__ float spw[12 * 24], sres[12 * 24];
  __shared__ float sinv2[12], soff2[12];

  int lx = threadIdx.x;                  // 0..27
  int y  = blockIdx.y * 4 + threadIdx.y; // 0..223
  int tx = blockIdx.x;                   // 0..7
  int ty = y / 28;
  int t  = ty * 8 + tx;
  int ly = y % 28;
  int b  = blockIdx.z;
  int tid = threadIdx.y * 28 + lx;       // 0..111

  for (int i = tid; i < 24 * 9; i += 112) sdw[i] = wdw[t * 216 + i];
  for (int i = tid; i < 24; i += 112) {
    float inv = g1[t * 24 + i] * rsqrtf(v1[t * 24 + i] + BN_EPS);
    sinv1[i] = inv;
    soff1[i] = b1[t * 24 + i] - m1[t * 24 + i] * inv;
  }
  for (int i = tid; i < 12 * 24; i += 112) {
    spw[i]  = wpw[t * 288 + i];
    sres[i] = wres[t * 288 + i];
  }
  for (int i = tid; i < 12; i += 112) {
    float inv = g2[t * 12 + i] * rsqrtf(v2[t * 12 + i] + BN_EPS);
    sinv2[i] = inv;
    soff2[i] = b2[t * 12 + i] - m2[t * 12 + i] * inv;
  }
  __syncthreads();

  int xg = tx * 28 + lx;
  float a[24], cen[24];
  for (int ci = 0; ci < 24; ++ci) {
    const float* base = gh + ((size_t)b * 24 + ci) * 224 * 224;
    float s = 0.f;
    #pragma unroll
    for (int dy = -1; dy <= 1; ++dy) {
      int lyy = ly + dy;
      if (lyy < 0 || lyy >= 28) continue;
      int yy = ty * 28 + lyy;
      #pragma unroll
      for (int dx = -1; dx <= 1; ++dx) {
        int lxx = lx + dx;
        if (lxx < 0 || lxx >= 28) continue;
        float v = base[(size_t)yy * 224 + tx * 28 + lxx];
        s += v * sdw[ci * 9 + (dy + 1) * 3 + (dx + 1)];
        if (dy == 0 && dx == 0) cen[ci] = v;
      }
    }
    a[ci] = geluf(s * sinv1[ci] + soff1[ci]);
  }

  size_t m = (size_t)b * 3136 + (y >> 2) * 56 + (xg >> 2);
  int colbase = (y & 3) * 4 + (xg & 3);

  for (int co = 0; co < 12; ++co) {
    float pw = 0.f, rs = 0.f;
    #pragma unroll
    for (int ci = 0; ci < 24; ++ci) {
      pw += a[ci] * spw[co * 24 + ci];
      rs += cen[ci] * sres[co * 24 + ci];
    }
    float g = geluf(pw * sinv2[co] + soff2[co] + rs);
    gb1[(((size_t)b * 12 + co) * 224 + y) * 224 + xg] = g;
    A1[m * 768 + (24 + co) * 16 + colbase] = (__bf16)g;
  }
}

// ---------------------------------------------------------------------------
// Branch2: 7x7 grid of 32x32 tiles, cin=cout=12, identity residual.
// Writes ONLY the A1 slice (channels 36..47). block (32,4), grid (7,56,32)
// ---------------------------------------------------------------------------
__global__ void branch2_kernel(const float* __restrict__ gb1,
                               const float* __restrict__ wdw,   // [49,12,9]
                               const float* __restrict__ g1, const float* __restrict__ b1,
                               const float* __restrict__ m1, const float* __restrict__ v1,
                               const float* __restrict__ wpw,   // [49,12,12]
                               const float* __restrict__ g2, const float* __restrict__ b2,
                               const float* __restrict__ m2, const float* __restrict__ v2,
                               __bf16* __restrict__ A1) {
  __shared__ float sdw[12 * 9];
  __shared__ float sinv1[12], soff1[12];
  __shared__ float spw[12 * 12];
  __shared__ float sinv2[12], soff2[12];

  int lx = threadIdx.x;                  // 0..31
  int y  = blockIdx.y * 4 + threadIdx.y;
  int tx = blockIdx.x;                   // 0..6
  int ty = y / 32;
  int t  = ty * 7 + tx;
  int ly = y % 32;
  int b  = blockIdx.z;
  int tid = threadIdx.y * 32 + lx;       // 0..127

  for (int i = tid; i < 12 * 9; i += 128) sdw[i] = wdw[t * 108 + i];
  for (int i = tid; i < 12; i += 128) {
    float inv = g1[t * 12 + i] * rsqrtf(v1[t * 12 + i] + BN_EPS);
    sinv1[i] = inv;
    soff1[i] = b1[t * 12 + i] - m1[t * 12 + i] * inv;
  }
  for (int i = tid; i < 12 * 12; i += 128) spw[i] = wpw[t * 144 + i];
  for (int i = tid; i < 12; i += 128) {
    float inv = g2[t * 12 + i] * rsqrtf(v2[t * 12 + i] + BN_EPS);
    sinv2[i] = inv;
    soff2[i] = b2[t * 12 + i] - m2[t * 12 + i] * inv;
  }
  __syncthreads();

  int xg = tx * 32 + lx;
  float a[12], cen[12];
  for (int ci = 0; ci < 12; ++ci) {
    const float* base = gb1 + ((size_t)b * 12 + ci) * 224 * 224;
    float s = 0.f;
    #pragma unroll
    for (int dy = -1; dy <= 1; ++dy) {
      int lyy = ly + dy;
      if (lyy < 0 || lyy >= 32) continue;
      int yy = ty * 32 + lyy;
      #pragma unroll
      for (int dx = -1; dx <= 1; ++dx) {
        int lxx = lx + dx;
        if (lxx < 0 || lxx >= 32) continue;
        float v = base[(size_t)yy * 224 + tx * 32 + lxx];
        s += v * sdw[ci * 9 + (dy + 1) * 3 + (dx + 1)];
        if (dy == 0 && dx == 0) cen[ci] = v;
      }
    }
    a[ci] = geluf(s * sinv1[ci] + soff1[ci]);
  }

  size_t m = (size_t)b * 3136 + (y >> 2) * 56 + (xg >> 2);
  int colbase = (y & 3) * 4 + (xg & 3);

  for (int co = 0; co < 12; ++co) {
    float pw = 0.f;
    #pragma unroll
    for (int ci = 0; ci < 12; ++ci) pw += a[ci] * spw[co * 12 + ci];
    float g = geluf(pw * sinv2[co] + soff2[co] + cen[co]);
    A1[m * 768 + (36 + co) * 16 + colbase] = (__bf16)g;
  }
}

// ---------------------------------------------------------------------------
// WMMA bf16 GEMM: C[M][N] = A[M][K=768] * B[N][K]^T, fused BN(+GELU) epilogue.
// Block = 256 threads = 8 waves; wave tile 16(M)x64(N); block tile 128x64.
// Whole 64x768 B panel staged once into LDS (padded stride 784 -> bank-
// conflict-free 32B fragment reads), then barrier-free K loop.
// EPI 0: gelu -> bf16 scatter into A2 ; EPI 1: gelu -> bf16 scatter into A3 ;
// EPI 2: plain -> fp32 d_out (contiguous).
// ---------------------------------------------------------------------------
template <int EPI>
__global__ void gemm_wmma_kernel(const __bf16* __restrict__ A,
                                 const __bf16* __restrict__ Bm,
                                 const float* __restrict__ bg,
                                 const float* __restrict__ bb,
                                 const float* __restrict__ bm,
                                 const float* __restrict__ pv,
                                 void* __restrict__ outp) {
  constexpr int K = 768;
  constexpr int LDB = 784;  // padded LDS row stride (halves)
  extern __shared__ __align__(32) __bf16 lB[];  // 64 * LDB halves = 98 KB

  int lane  = threadIdx.x & 31;
  int wave  = threadIdx.x >> 5;
  int mrow  = blockIdx.x * 128 + wave * 16;
  int nBase = blockIdx.y * 64;

  // ---- stage whole B panel: 64 rows x 96 chunks of 8 halves (16B) ----
  for (int c = threadIdx.x; c < 64 * 96; c += 256) {
    int n = c / 96, k8 = (c % 96) * 8;
    const __bf16* src = Bm + (size_t)(nBase + n) * K + k8;
    __bf16* dst = lB + n * LDB + k8;
#if USE_ASYNC_LDS
    __builtin_amdgcn_global_load_async_to_lds_b128(
        (AS1 v4i*)(AS1 void*)const_cast<__bf16*>(src),
        (AS3 v4i*)(AS3 void*)dst, 0, 0);
#else
    *(v8bf*)dst = *(const v8bf*)src;
#endif
  }
#if USE_ASYNC_LDS
  wait_async0();
#endif
  __syncthreads();

  v8f acc[4];
  #pragma unroll
  for (int f = 0; f < 4; ++f)
    #pragma unroll
    for (int j = 0; j < 8; ++j) acc[f][j] = 0.f;

  // BN params for this lane's 4 output columns
  float ginv[4], goff[4];
  #pragma unroll
  for (int f = 0; f < 4; ++f) {
    int n = nBase + f * 16 + (lane & 15);
    float inv = bg[n] * rsqrtf(pv[n] + BN_EPS);
    ginv[f] = inv;
    goff[f] = bb[n] - bm[n] * inv;
  }

  // A fragment (16-bit A 16x32, this code's k order, not the ISA table): lane holds row (lane&15);
  // halves 0..15 = contiguous K[(lane>>4)*16 .. +15].
  const __bf16* arow = A + (size_t)(mrow + (lane & 15)) * K + ((lane >> 4) << 4);
  // B fragment (B 32x16, same k order as A): lane holds col (lane&15);
  // halves 0..15 = contiguous K[(lane>>4)*16 .. +15].
  const __bf16* bbase = lB + (size_t)(lane & 15) * LDB + ((lane >> 4) << 4);

  #pragma unroll 4
  for (int kt = 0; kt < K; kt += 32) {
    v8bf a0 = *(const v8bf*)(arow + kt);
    v8bf a1 = *(const v8bf*)(arow + kt + 8);
    v16bf av = __builtin_shufflevector(a0, a1, 0, 1, 2, 3, 4, 5, 6, 7,
                                               8, 9, 10, 11, 12, 13, 14, 15);
    #pragma unroll
    for (int f = 0; f < 4; ++f) {
      v16bf bfrag = *(const v16bf*)(bbase + f * 16 * LDB + kt);
      acc[f] = __builtin_amdgcn_wmma_f32_16x16x32_bf16(
          false, av, false, bfrag, (short)0, acc[f], false, false);
    }
  }

  // Epilogue: C/D layout — VGPR j holds M = mrow + j + (lane>=16 ? 8 : 0),
  // N = nBase + f*16 + (lane&15).
  #pragma unroll
  for (int f = 0; f < 4; ++f) {
    int N_ = nBase + f * 16 + (lane & 15);
    #pragma unroll
    for (int j = 0; j < 8; ++j) {
      int M_ = mrow + j + ((lane >> 4) << 3);
      float y = acc[f][j] * ginv[f] + goff[f];
      if (EPI == 0) {
        int b = M_ / 3136, r = M_ % 3136;
        int oh = r / 56, ow = r % 56;
        int row = b * 784 + (oh >> 1) * 28 + (ow >> 1);
        int col = N_ * 4 + (oh & 1) * 2 + (ow & 1);
        ((__bf16*)outp)[(size_t)row * 768 + col] = (__bf16)geluf(y);
      } else if (EPI == 1) {
        int b = M_ / 784, r = M_ % 784;
        int oh = r / 28, ow = r % 28;
        int row = b * 196 + (oh >> 1) * 14 + (ow >> 1);
        int col = N_ * 4 + (oh & 1) * 2 + (ow & 1);
        ((__bf16*)outp)[(size_t)row * 768 + col] = (__bf16)geluf(y);
      } else {
        ((float*)outp)[(size_t)M_ * 768 + N_] = y;
      }
    }
  }
}

// ---------------------------------------------------------------------------
// Host launcher. Input order assumes insertion-order flattening of the
// setup_inputs() nested dicts:
//  0:x 1:w_first 2-5:bn_first(g,b,m,v)
//  6:b1.w_dw 7-10:b1.bn1 11:b1.w_pw 12-15:b1.bn2 16:b1.w_res
//  17:b2.w_dw 18-21:b2.bn1 22:b2.w_pw 23-26:b2.bn2
//  27:w1 28-31:bn1 32:w2 33-36:bn2 37:w3 38-41:bn3
// ---------------------------------------------------------------------------
extern "C" void kernel_launch(void* const* d_in, const int* in_sizes, int n_in,
                              void* d_out, int out_size, void* d_ws, size_t ws_size,
                              hipStream_t stream) {
  const float* x       = (const float*)d_in[0];
  const float* w_first = (const float*)d_in[1];
  const float* bf_g = (const float*)d_in[2];
  const float* bf_b = (const float*)d_in[3];
  const float* bf_m = (const float*)d_in[4];
  const float* bf_v = (const float*)d_in[5];

  const float* b1_wdw = (const float*)d_in[6];
  const float* b1_g1  = (const float*)d_in[7];
  const float* b1_b1  = (const float*)d_in[8];
  const float* b1_m1  = (const float*)d_in[9];
  const float* b1_v1  = (const float*)d_in[10];
  const float* b1_wpw = (const float*)d_in[11];
  const float* b1_g2  = (const float*)d_in[12];
  const float* b1_b2  = (const float*)d_in[13];
  const float* b1_m2  = (const float*)d_in[14];
  const float* b1_v2  = (const float*)d_in[15];
  const float* b1_wres= (const float*)d_in[16];

  const float* b2_wdw = (const float*)d_in[17];
  const float* b2_g1  = (const float*)d_in[18];
  const float* b2_b1  = (const float*)d_in[19];
  const float* b2_m1  = (const float*)d_in[20];
  const float* b2_v1  = (const float*)d_in[21];
  const float* b2_wpw = (const float*)d_in[22];
  const float* b2_g2  = (const float*)d_in[23];
  const float* b2_b2  = (const float*)d_in[24];
  const float* b2_m2  = (const float*)d_in[25];
  const float* b2_v2  = (const float*)d_in[26];

  const float* w1  = (const float*)d_in[27];
  const float* n1g = (const float*)d_in[28];
  const float* n1b = (const float*)d_in[29];
  const float* n1m = (const float*)d_in[30];
  const float* n1v = (const float*)d_in[31];
  const float* w2  = (const float*)d_in[32];
  const float* n2g = (const float*)d_in[33];
  const float* n2b = (const float*)d_in[34];
  const float* n2m = (const float*)d_in[35];
  const float* n2v = (const float*)d_in[36];
  const float* w3  = (const float*)d_in[37];
  const float* n3g = (const float*)d_in[38];
  const float* n3b = (const float*)d_in[39];
  const float* n3m = (const float*)d_in[40];
  const float* n3v = (const float*)d_in[41];

  // Workspace carve-out
  char* ws = (char*)d_ws;
  size_t off = 0;
  auto carve = [&](size_t bytes) -> void* {
    void* p = ws + off;
    off += (bytes + 255) & ~(size_t)255;
    return p;
  };
  float*  gh  = (float*)carve((size_t)32 * 24 * 224 * 224 * 4); // 154.1 MB
  float*  gb1 = (float*)carve((size_t)32 * 12 * 224 * 224 * 4); //  77.1 MB
  __bf16* A1  = (__bf16*)carve((size_t)100352 * 768 * 2);       // 154.1 MB
  __bf16* A2  = (__bf16*)carve((size_t)25088 * 768 * 2);        //  38.5 MB
  __bf16* A3  = (__bf16*)carve((size_t)6272 * 768 * 2);         //   9.6 MB
  __bf16* W1b = (__bf16*)carve((size_t)192 * 768 * 2);
  __bf16* W2b = (__bf16*)carve((size_t)192 * 768 * 2);
  __bf16* W3b = (__bf16*)carve((size_t)768 * 768 * 2);
  if (off > ws_size) return;  // insufficient scratch; deterministic no-op

  const size_t ldsB = (size_t)64 * 784 * 2;  // 98 KB dynamic LDS per block

  // 1) weight conversion (layouts already match GEMM [N][K])
  cvt_bf16_kernel<<<dim3((147456 + 255) / 256), dim3(256), 0, stream>>>(w1, W1b, 147456);
  cvt_bf16_kernel<<<dim3((147456 + 255) / 256), dim3(256), 0, stream>>>(w2, W2b, 147456);
  cvt_bf16_kernel<<<dim3((589824 + 255) / 256), dim3(256), 0, stream>>>(w3, W3b, 589824);

  // 2) first conv + BN + GELU -> gh, A1[:, 0:384]
  first_conv_kernel<<<dim3(7, 28, 32), dim3(32, 8), 0, stream>>>(
      x, w_first, bf_g, bf_b, bf_m, bf_v, gh, A1);

  // 3) region branches (fused im2col writes)
  branch1_kernel<<<dim3(8, 56, 32), dim3(28, 4), 0, stream>>>(
      gh, b1_wdw, b1_g1, b1_b1, b1_m1, b1_v1,
      b1_wpw, b1_g2, b1_b2, b1_m2, b1_v2, b1_wres, gb1, A1);
  branch2_kernel<<<dim3(7, 56, 32), dim3(32, 4), 0, stream>>>(
      gb1, b2_wdw, b2_g1, b2_b1, b2_m1, b2_v1,
      b2_wpw, b2_g2, b2_b2, b2_m2, b2_v2, A1);

  // 4) WMMA GEMM stack
  // conv1: M=100352 (784 blocks of 128), N=192 (3 tiles of 64), K=768
  gemm_wmma_kernel<0><<<dim3(784, 3), dim3(256), ldsB, stream>>>(
      A1, W1b, n1g, n1b, n1m, n1v, (void*)A2);
  // conv2: M=25088 (196), N=192 (3), K=768
  gemm_wmma_kernel<1><<<dim3(196, 3), dim3(256), ldsB, stream>>>(
      A2, W2b, n2g, n2b, n2m, n2v, (void*)A3);
  // conv3: M=6272 (49), N=768 (12), K=768 -> fp32 output [B,196,768]
  gemm_wmma_kernel<2><<<dim3(49, 12), dim3(256), ldsB, stream>>>(
      A3, W3b, n3g, n3b, n3m, n3v, d_out);

  (void)in_sizes; (void)n_in; (void)out_size;
}
